// SPR_RGCN_88648124990250
// MI455X (gfx1250) — matrix operand order repaired and checked by reading, not compiled, not run on hardware
//
#include <hip/hip_runtime.h>
#include <hip/hip_bf16.h>

// ---------------------------------------------------------------------------
// Problem constants (match reference)
// ---------------------------------------------------------------------------
#define N_NODES  50000
#define N_EDGES  600000
#define N_GRAPHS 64
#define EMB      32
#define HID      128
#define N_REL    3
#define MAX_POS  25

typedef __attribute__((ext_vector_type(16))) __bf16 v16bf;
typedef __attribute__((ext_vector_type(8)))  float  v8f;

// ---------------------------------------------------------------------------
// 1. Embedding concat: h0[i] = [shape_emb[x0], color_emb[x1], pos_emb[clip(x2)]]
// ---------------------------------------------------------------------------
__global__ void embed_kernel(const int* __restrict__ x,
                             const float* __restrict__ se,
                             const float* __restrict__ ce,
                             const float* __restrict__ pe,
                             float* __restrict__ h) {
    size_t idx = (size_t)blockIdx.x * blockDim.x + threadIdx.x;
    if (idx >= (size_t)N_NODES * 96) return;
    int node = (int)(idx / 96);
    int f = (int)(idx % 96);
    float v;
    if (f < 32) {
        v = se[x[node * 3 + 0] * EMB + f];
    } else if (f < 64) {
        v = ce[x[node * 3 + 1] * EMB + (f - 32)];
    } else {
        int p = x[node * 3 + 2];
        p = p < 0 ? 0 : (p > MAX_POS - 1 ? MAX_POS - 1 : p);
        v = pe[p * EMB + (f - 64)];
    }
    h[idx] = v;
}

// ---------------------------------------------------------------------------
// 2. Edge scatter-add: pre[r][dst] += h[src]; cnt[r][dst] += 1
//    one thread per (edge, 4-float chunk)
// ---------------------------------------------------------------------------
__global__ void edge_agg_kernel(const int* __restrict__ ei,
                                const int* __restrict__ et,
                                const float* __restrict__ h,
                                float* __restrict__ pre,
                                float* __restrict__ cnt,
                                int Din) {
    int chunks = Din >> 2;
    size_t idx = (size_t)blockIdx.x * blockDim.x + threadIdx.x;
    if (idx >= (size_t)N_EDGES * chunks) return;
    int e = (int)(idx / chunks);
    int c = (int)(idx % chunks);
    int s = ei[e];
    int d = ei[N_EDGES + e];
    int r = et[e];
    float4 hv = reinterpret_cast<const float4*>(h + (size_t)s * Din)[c];
    float* p = pre + ((size_t)r * N_NODES + d) * Din + c * 4;
    atomicAdd(p + 0, hv.x);
    atomicAdd(p + 1, hv.y);
    atomicAdd(p + 2, hv.z);
    atomicAdd(p + 3, hv.w);
    if (c == 0) atomicAdd(&cnt[(size_t)r * N_NODES + d], 1.0f);
}

// ---------------------------------------------------------------------------
// 3. Build Xcat (bf16): [h | pre0/c0 | pre1/c1 | pre2/c2]  row-major, KCAT=4*Din
// ---------------------------------------------------------------------------
__global__ void build_xcat_kernel(const float* __restrict__ h,
                                  const float* __restrict__ pre,
                                  const float* __restrict__ cnt,
                                  __bf16* __restrict__ X,
                                  int Din) {
    int KCAT = 4 * Din;
    size_t idx = (size_t)blockIdx.x * blockDim.x + threadIdx.x;
    if (idx >= (size_t)N_NODES * KCAT) return;
    int node = (int)(idx / KCAT);
    int k = (int)(idx % KCAT);
    float v;
    if (k < Din) {
        v = h[(size_t)node * Din + k];
    } else {
        int r = k / Din - 1;
        int kk = k % Din;
        float c = cnt[(size_t)r * N_NODES + node];
        v = pre[((size_t)r * N_NODES + node) * Din + kk] / fmaxf(c, 1.0f);
    }
    X[idx] = (__bf16)v;
}

// ---------------------------------------------------------------------------
// 4. Pack Wcat = [root; W0; W1; W2] into WMMA B-fragment layout (bf16).
//    B tile 32x16, contiguous k order (same as A): lane = (krel>>4)*16 + (n&15); slot = 2*((krel&15)>>1) + (krel&1)
//    Fragment storage: ((kt*8 + nt)*32 + lane)*16 + slot   (contiguous 32B/lane)
// ---------------------------------------------------------------------------
__global__ void pack_w_kernel(const float* __restrict__ root,
                              const float* __restrict__ W,
                              __bf16* __restrict__ Wp,
                              int Din) {
    int KCAT = 4 * Din;
    int idx = blockIdx.x * blockDim.x + threadIdx.x;
    if (idx >= KCAT * HID) return;
    int k = idx >> 7;      // 0..KCAT-1
    int n = idx & 127;
    float v;
    if (k < Din) {
        v = root[k * HID + n];
    } else {
        int r = k / Din - 1;
        int kk = k % Din;
        v = W[((size_t)r * Din + kk) * HID + n];
    }
    int kt   = k >> 5;
    int krel = k & 31;
    int nt   = n >> 4;
    int coln = n & 15;
    int hf   = krel >> 4;
    int vr   = (krel & 15) >> 1;
    int par  = krel & 1;
    int lane = hf * 16 + coln;
    size_t off = (((size_t)(kt * 8 + nt)) * 32 + lane) * 16 + vr * 2 + par;
    Wp[off] = (__bf16)v;
}

// ---------------------------------------------------------------------------
// 5. WMMA GEMM + bias + ReLU:  Y(50000x128) = relu(X(50000xKCAT) @ Wcat + b)
//    8 waves/block; 1 wave = one 16-row strip, all 128 output cols.
//    B staged via LDS in 128-deep K chunks (32KB), fragment-ready.
// ---------------------------------------------------------------------------
template <int KCAT>
__global__ __launch_bounds__(256) void gemm_bias_relu_kernel(
        const __bf16* __restrict__ X,
        const __bf16* __restrict__ Wp,
        const float*  __restrict__ bias,
        float* __restrict__ Y) {
    constexpr int KCHUNK = 128;
    constexpr int CHUNK_U4 = KCHUNK * HID * 2 / 16;   // 2048 uint4 = 32KB
    __shared__ uint4 ldsB[CHUNK_U4];

    const int tid  = threadIdx.x;
    const int wave = tid >> 5;
    const int lane = tid & 31;
    const int half = lane >> 4;
    const int col  = lane & 15;
    const int strip = blockIdx.x * 8 + wave;
    const bool active = (strip * 16) < N_NODES;   // wave-uniform
    const int row0 = strip * 16;

    v8f acc[8] = {};
    float bc[8];
#pragma unroll
    for (int nt = 0; nt < 8; ++nt) bc[nt] = bias[nt * 16 + col];

    for (int kc = 0; kc < KCAT / KCHUNK; ++kc) {
        __syncthreads();
        const uint4* src = reinterpret_cast<const uint4*>(Wp) + (size_t)kc * CHUNK_U4;
#pragma unroll
        for (int i = 0; i < CHUNK_U4 / 256; ++i)
            ldsB[tid + i * 256] = src[tid + i * 256];
        __syncthreads();

        if (active) {
#pragma unroll
            for (int kt = 0; kt < KCHUNK / 32; ++kt) {
                // A fragment: row = row0 + (lane&15); K = kbase + half*16 .. +15
                const int kbase = kc * KCHUNK + kt * 32;
                union { uint4 u[2]; v16bf v; } A;
                const uint4* Ap = reinterpret_cast<const uint4*>(
                    X + (size_t)(row0 + col) * KCAT + kbase + half * 16);
                A.u[0] = Ap[0];   // K = kbase + half*16 .. +7
                A.u[1] = Ap[1];   // K = kbase + half*16 + 8 .. +15
#pragma unroll
                for (int nt = 0; nt < 8; ++nt) {
                    union { uint4 u[2]; v16bf v; } B;
                    const uint4* Bp = &ldsB[(((kt * 8 + nt) * 32) + lane) * 2];
                    B.u[0] = Bp[0];
                    B.u[1] = Bp[1];
                    acc[nt] = __builtin_amdgcn_wmma_f32_16x16x32_bf16(
                        false, A.v, false, B.v, (short)0, acc[nt], false, false);
                }
            }
        }
    }

    if (active) {
        // C layout: VGPR i -> row = row0 + half*8 + i, col = nt*16 + (lane&15)
#pragma unroll
        for (int nt = 0; nt < 8; ++nt) {
#pragma unroll
            for (int i = 0; i < 8; ++i) {
                int r = row0 + half * 8 + i;
                if (r < N_NODES) {
                    float v = acc[nt][i] + bc[nt];
                    Y[(size_t)r * HID + nt * 16 + col] = v > 0.0f ? v : 0.0f;
                }
            }
        }
    }
}

// ---------------------------------------------------------------------------
// 6. Mean pool per graph (scatter-add) + counts
// ---------------------------------------------------------------------------
__global__ void pool_kernel(const float* __restrict__ h,
                            const int* __restrict__ batch,
                            float* __restrict__ pooled,
                            float* __restrict__ cntg) {
    size_t idx = (size_t)blockIdx.x * blockDim.x + threadIdx.x;
    if (idx >= (size_t)N_NODES * HID) return;
    int node = (int)(idx >> 7);
    int c = (int)(idx & 127);
    int g = batch[node];
    atomicAdd(&pooled[(size_t)g * HID + c], h[idx]);
    if (c == 0) atomicAdd(&cntg[g], 1.0f);
}

// ---------------------------------------------------------------------------
// 7. Classifier: out(64x10) = (pooled/cnt) @ cls_w + cls_b
// ---------------------------------------------------------------------------
__global__ void cls_kernel(const float* __restrict__ pooled,
                           const float* __restrict__ cntg,
                           const float* __restrict__ w,
                           const float* __restrict__ b,
                           float* __restrict__ out) {
    int idx = blockIdx.x * blockDim.x + threadIdx.x;
    if (idx >= N_GRAPHS * 10) return;
    int g = idx / 10;
    int j = idx % 10;
    float inv = 1.0f / fmaxf(cntg[g], 1.0f);
    float acc = b[j];
    for (int c = 0; c < HID; ++c)
        acc += pooled[(size_t)g * HID + c] * inv * w[c * 10 + j];
    out[idx] = acc;
}

// ---------------------------------------------------------------------------
// Host launcher
// ---------------------------------------------------------------------------
extern "C" void kernel_launch(void* const* d_in, const int* in_sizes, int n_in,
                              void* d_out, int out_size, void* d_ws, size_t ws_size,
                              hipStream_t stream) {
    const int*   x     = (const int*)d_in[0];
    const int*   ei    = (const int*)d_in[1];
    const int*   et    = (const int*)d_in[2];
    const int*   batch = (const int*)d_in[3];
    const float* se    = (const float*)d_in[4];
    const float* ce    = (const float*)d_in[5];
    const float* pe    = (const float*)d_in[6];
    const float* W1    = (const float*)d_in[7];
    const float* root1 = (const float*)d_in[8];
    const float* b1    = (const float*)d_in[9];
    const float* W2    = (const float*)d_in[10];
    const float* root2 = (const float*)d_in[11];
    const float* b2    = (const float*)d_in[12];
    const float* clsw  = (const float*)d_in[13];
    const float* clsb  = (const float*)d_in[14];
    float* out = (float*)d_out;

    // workspace layout (bytes, 16B-aligned partitions)
    char* ws = (char*)d_ws;
    constexpr size_t SZ_H0    = (size_t)N_NODES * 96 * 4;         // 19.2 MB
    constexpr size_t SZ_H     = (size_t)N_NODES * HID * 4;        // 25.6 MB
    constexpr size_t SZ_PRE   = (size_t)N_REL * N_NODES * HID * 4;// 76.8 MB
    constexpr size_t SZ_CNT   = (size_t)N_REL * N_NODES * 4;      // 0.6 MB
    constexpr size_t SZ_XCAT  = (size_t)N_NODES * 512 * 2;        // 51.2 MB
    constexpr size_t SZ_WPACK = (size_t)512 * HID * 2;            // 128 KB
    constexpr size_t SZ_POOL  = (size_t)N_GRAPHS * HID * 4;
    constexpr size_t SZ_CNTG  = (size_t)N_GRAPHS * 4;

    size_t off = 0;
    float*  h0     = (float*)(ws + off);  off += SZ_H0;
    float*  h1     = (float*)(ws + off);  off += SZ_H;
    float*  h2     = (float*)(ws + off);  off += SZ_H;
    float*  pre    = (float*)(ws + off);  off += SZ_PRE;
    float*  cnt    = (float*)(ws + off);  off += SZ_CNT;
    __bf16* xcat   = (__bf16*)(ws + off); off += SZ_XCAT;
    __bf16* wpack  = (__bf16*)(ws + off); off += SZ_WPACK;
    float*  pooled = (float*)(ws + off);  off += SZ_POOL;
    float*  cntg   = (float*)(ws + off);  off += SZ_CNTG;

    const int T = 256;
    const int gemm_blocks = ((N_NODES / 16) + 7) / 8;   // 3125 strips / 8 waves

    // ---- node features
    embed_kernel<<<(int)(((size_t)N_NODES * 96 + T - 1) / T), T, 0, stream>>>(
        x, se, ce, pe, h0);

    // ---- layer 1 (Din=96, KCAT=384)
    hipMemsetAsync(pre, 0, SZ_PRE + SZ_CNT, stream);    // pre & cnt contiguous
    edge_agg_kernel<<<(int)(((size_t)N_EDGES * 24 + T - 1) / T), T, 0, stream>>>(
        ei, et, h0, pre, cnt, 96);
    build_xcat_kernel<<<(int)(((size_t)N_NODES * 384 + T - 1) / T), T, 0, stream>>>(
        h0, pre, cnt, xcat, 96);
    pack_w_kernel<<<(384 * HID + T - 1) / T, T, 0, stream>>>(root1, W1, wpack, 96);
    gemm_bias_relu_kernel<384><<<gemm_blocks, T, 0, stream>>>(xcat, wpack, b1, h1);

    // ---- layer 2 (Din=128, KCAT=512)
    hipMemsetAsync(pre, 0, SZ_PRE + SZ_CNT, stream);
    edge_agg_kernel<<<(int)(((size_t)N_EDGES * 32 + T - 1) / T), T, 0, stream>>>(
        ei, et, h1, pre, cnt, 128);
    build_xcat_kernel<<<(int)(((size_t)N_NODES * 512 + T - 1) / T), T, 0, stream>>>(
        h1, pre, cnt, xcat, 128);
    pack_w_kernel<<<(512 * HID + T - 1) / T, T, 0, stream>>>(root2, W2, wpack, 128);
    gemm_bias_relu_kernel<512><<<gemm_blocks, T, 0, stream>>>(xcat, wpack, b2, h2);

    // ---- pool + classify
    hipMemsetAsync(pooled, 0, SZ_POOL + SZ_CNTG, stream);
    pool_kernel<<<(int)(((size_t)N_NODES * HID + T - 1) / T), T, 0, stream>>>(
        h2, batch, pooled, cntg);
    cls_kernel<<<(N_GRAPHS * 10 + T - 1) / T, T, 0, stream>>>(
        pooled, cntg, clsw, clsb, out);
}
